// ALICNN_84670985273441
// MI455X (gfx1250) — matrix operand order repaired and checked by reading, not compiled, not run on hardware
//
#include <hip/hip_runtime.h>

typedef __attribute__((ext_vector_type(16))) _Float16 v16h;
typedef __attribute__((ext_vector_type(8)))  float    v8f;

#define IMG_W 512
#define IMG_H 512
#define OTW   64      // output tile width
#define OTH   32      // output tile height
#define THALO 38      // OTH + 6 halo rows
#define TW    70      // OTW + 6 halo cols
#define TS    72      // LDS tile row stride (floats)
#define RS    64      // rowsum row stride

// ---------------------------------------------------------------- scalars init
__global__ void alicnn_zero_ws(float* ws) {
    if (threadIdx.x < 2) ws[threadIdx.x] = 0.0f;
}

// ---------------------------------------------------------------- pass A: sum(x^2)
__global__ __launch_bounds__(256) void alicnn_sumsq(const float4* __restrict__ x4,
                                                    int n4, float* __restrict__ ws) {
    float acc = 0.0f;
    int stride = gridDim.x * blockDim.x;
    for (int i = blockIdx.x * blockDim.x + threadIdx.x; i < n4; i += stride) {
        float4 v = x4[i];
        acc += v.x * v.x + v.y * v.y + v.z * v.z + v.w * v.w;
    }
    for (int m = 16; m >= 1; m >>= 1) acc += __shfl_xor(acc, m, 32);
    __shared__ float wsum[8];
    int lane = threadIdx.x & 31, wv = threadIdx.x >> 5;
    if (lane == 0) wsum[wv] = acc;
    __syncthreads();
    if (threadIdx.x == 0) {
        float s = 0.0f;
        #pragma unroll
        for (int i = 0; i < 8; ++i) s += wsum[i];
        atomicAdd(ws, s);
    }
}

// ---------------------------------------------------------------- pass B: stencil + WMMA
__global__ __launch_bounds__(256) void alicnn_main(const float* __restrict__ x,
                                                   float* __restrict__ supp_out,   // mask slot (temp)
                                                   float* __restrict__ avg_out,
                                                   float* __restrict__ diff_out,
                                                   float* __restrict__ ws) {
    __shared__ float    tile[THALO * TS];
    __shared__ float    rowsum[THALO * RS];
    __shared__ _Float16 kwh[64];     // mex-hat weights (0 for tap >= 49)
    __shared__ float    blockSS;

    const int tid = threadIdx.x;
    const int ty0 = blockIdx.y * OTH;
    const int tx0 = blockIdx.x * OTW;
    const size_t imgBase = (size_t)blockIdx.z * (IMG_W * IMG_H);

    const float rnorm = 1.0f / sqrtf(ws[0]);

    if (tid == 0) blockSS = 0.0f;
    if (tid < 64) {                                   // weight table (zero-padded to 64 taps)
        int t = tid;
        float w = 0.0f;
        if (t < 49) {
            int dy = t / 7, dx = t % 7;
            float fy = (float)(dy - 3), fx = (float)(dx - 3);
            float e = sqrtf(fy * fy + fx * fx) * (1.0f / 7.0f);
            w = e * __expf(-e);
        }
        kwh[t] = (_Float16)w;
    }
    // normalized, zero-padded halo tile
    for (int i = tid; i < THALO * TW; i += 256) {
        int yy = i / TW, xx = i - yy * TW;
        int gy = ty0 + yy - 3, gx = tx0 + xx - 3;
        float v = 0.0f;
        if (gy >= 0 && gy < IMG_H && gx >= 0 && gx < IMG_W)
            v = x[imgBase + (size_t)gy * IMG_W + gx] * rnorm;
        tile[yy * TS + xx] = v;
    }
    __syncthreads();
    // separable box filter, horizontal pass
    for (int i = tid; i < THALO * OTW; i += 256) {
        int yy = i >> 6, xx = i & 63;
        const float* tp = &tile[yy * TS + xx];
        rowsum[yy * RS + xx] = tp[0] + tp[1] + tp[2] + tp[3] + tp[4] + tp[5] + tp[6];
    }
    __syncthreads();

    const int wave = tid >> 5;
    const int lane = tid & 31;
    const int half = lane >> 4;       // 0: taps K 0..15 / 1: taps K 16..31 (B layout)
    const int lp   = lane & 15;       // pixel within 16-pixel group

    // Per-lane tap -> LDS offset map, fully resolved to registers (two compile-time
    // candidates per slot, one cndmask each at setup; padded taps alias offset 0 and
    // are annihilated by the zero A-weight).
    int offv[32];
    #pragma unroll
    for (int s = 0; s < 2; ++s) {
        #pragma unroll
        for (int e = 0; e < 16; ++e) {
            const int t0 = (s << 5) + e;          // half == 0
            const int t1 = t0 + 16;               // half == 1
            const int o0 = (t0 < 49) ? (t0 / 7) * TS + (t0 % 7) : 0;
            const int o1 = (t1 < 49) ? (t1 / 7) * TS + (t1 % 7) : 0;
            offv[(s << 4) + e] = half ? o1 : o0;
        }
    }
    // A fragments (weights replicated over rows), invariant over rows/groups.
    // A-layout K map: K = e + 16*half.
    v16h a0, a1;
    #pragma unroll
    for (int e = 0; e < 16; ++e) {
        const int kA = e + (half << 4);
        a0[e] = kwh[kA];
        a1[e] = kwh[32 + kA];
    }

    float ssAcc = 0.0f;

    for (int rr = 0; rr < 4; ++rr) {
        const int py = wave * 4 + rr;
        for (int g = 0; g < 4; ++g) {
            const int px   = g * 16 + lp;
            const int base = py * TS + px;
            const float xc = tile[base + 3 * TS + 3];

            v16h b;
            v8f  c = {0.f, 0.f, 0.f, 0.f, 0.f, 0.f, 0.f, 0.f};
            #pragma unroll
            for (int e = 0; e < 16; ++e)
                b[e] = (_Float16)fmaxf(tile[base + offv[e]] - xc, 0.0f);
            c = __builtin_amdgcn_wmma_f32_16x16x32_f16(false, a0, false, b,
                                                       (short)0, c, false, false);
            #pragma unroll
            for (int e = 0; e < 16; ++e)
                b[e] = (_Float16)fmaxf(tile[base + offv[16 + e]] - xc, 0.0f);
            c = __builtin_amdgcn_wmma_f32_16x16x32_f16(false, a1, false, b,
                                                       (short)0, c, false, false);

            const float diff = c[0];              // D rows replicated -> every lane has col lp

            float boxs = 0.0f;                    // vertical pass of box filter
            #pragma unroll
            for (int dy = 0; dy < 7; ++dy) boxs += rowsum[(py + dy) * RS + px];
            const float avg  = __expf(-boxs * (1.0f / 49.0f));
            const float supp = 0.1f * avg + 0.9f * diff;

            if (half == 0) {
                size_t idx = imgBase + (size_t)(ty0 + py) * IMG_W + (tx0 + px);
                avg_out[idx]  = avg;
                diff_out[idx] = diff;
                supp_out[idx] = supp;
                ssAcc += supp * supp;
            }
        }
    }
    for (int m = 16; m >= 1; m >>= 1) ssAcc += __shfl_xor(ssAcc, m, 32);
    if (lane == 0) atomicAdd(&blockSS, ssAcc);
    __syncthreads();
    if (tid == 0) atomicAdd(&ws[1], blockSS);
}

// ---------------------------------------------------------------- pass C: threshold mask
__global__ __launch_bounds__(256) void alicnn_mask(const float* __restrict__ x,
                                                   float* __restrict__ mask_io,
                                                   const float* __restrict__ ws, int n) {
    int i = blockIdx.x * blockDim.x + threadIdx.x;
    if (i >= n) return;
    float rs1 = 1.0f / sqrtf(ws[0]);
    float rs2 = 1.0f / sqrtf(ws[1]);
    float xn = x[i] * rs1;
    float sn = mask_io[i] * rs2;
    mask_io[i] = (xn > sn) ? 1.0f : 0.0f;
}

extern "C" void kernel_launch(void* const* d_in, const int* in_sizes, int n_in,
                              void* d_out, int out_size, void* d_ws, size_t ws_size,
                              hipStream_t stream) {
    const float* x = (const float*)d_in[0];
    const int n = in_sizes[0];                 // 8*512*512
    float* out      = (float*)d_out;
    float* out_mask = out;                     // also holds supp temporarily
    float* out_avg  = out + n;
    float* out_diff = out + 2 * (size_t)n;
    float* ws = (float*)d_ws;                  // ws[0]=sum(x^2), ws[1]=sum(supp^2)

    alicnn_zero_ws<<<1, 32, 0, stream>>>(ws);
    alicnn_sumsq<<<512, 256, 0, stream>>>((const float4*)x, n / 4, ws);

    dim3 grid(IMG_W / OTW, IMG_H / OTH, n / (IMG_W * IMG_H));   // (8, 16, 8)
    alicnn_main<<<grid, 256, 0, stream>>>(x, out_mask, out_avg, out_diff, ws);

    alicnn_mask<<<(n + 255) / 256, 256, 0, stream>>>(x, out_mask, ws, n);
}
